// PrototypeContrastiveLoss_25116968747302
// MI455X (gfx1250) — matrix operand order repaired and checked by reading, compiled, not run on hardware
//
#include <hip/hip_runtime.h>
#include <hip/hip_bf16.h>

// ---------------------------------------------------------------------------
// PrototypeContrastiveLoss on MI455X (gfx1250, wave32, WMMA + async-to-LDS)
//
// B=128, K=256, D=256, M=4, P=1024. GEMM (32768 x 1024 x 256) via
// v_wmma_f32_16x16x32_f16. Prototype tiles are staged cooperatively into
// LDS with GLOBAL_LOAD_ASYNC_TO_LDS_B128 (double-buffered, ASYNCcnt), so
// each block reads the 512KB prototype matrix once instead of once per
// wave (8x less L2 traffic -> compute-bound). All LDS traffic goes through
// explicit addrspace(3) pointers so the backend emits ds_load_b128 (not
// flat_load). Streaming sum-of-exp needs no max shift (|sims/T| <= ~10).
// ---------------------------------------------------------------------------

typedef _Float16 v8h  __attribute__((ext_vector_type(8)));
typedef _Float16 v16h __attribute__((ext_vector_type(16)));
typedef float    v8f  __attribute__((ext_vector_type(8)));
typedef int      v4i  __attribute__((vector_size(16)));   // matches builtin param

#define KC 256          // concepts
#define DD 256          // feature dim
#define MM 4            // prototypes per concept
#define PP (KC * MM)    // 1024 prototypes
#define NQ (128 * KC)   // 32768 query rows (B*K)
#define NT (PP / 16)    // 64 prototype column tiles
#define BSTR 272        // padded LDS row stride in halves (544B: kills bank conflicts)
#define BUFH (16 * BSTR) // halves per B buffer
#define TEMP_INV 10.0f
#define NORM_EPS 1e-12f

#define AS1 __attribute__((address_space(1)))
#define AS3 __attribute__((address_space(3)))

#if __has_builtin(__builtin_amdgcn_global_load_async_to_lds_b128) && \
    __has_builtin(__builtin_amdgcn_s_wait_asynccnt)
#define HAVE_ASYNC 1
#else
#define HAVE_ASYNC 0
#endif

__device__ __forceinline__ float waveReduceSum32(float v) {
#pragma unroll
    for (int off = 16; off > 0; off >>= 1)
        v += __shfl_xor(v, off, 32);
    return v;
}

// Stage one 16-prototype tile (16 rows x 256 halves) into LDS at lbase,
// padded row stride BSTR halves. 512 chunks of 16B; 2 per thread.
// gbase/lbase are pre-cast AS1/AS3 pointers (casts hoisted out of the loop).
__device__ __forceinline__ void stage_tile(AS1 const char* gbase,
                                           AS3 char* lbase, int t, int tid) {
#pragma unroll
    for (int j = 0; j < 2; ++j) {
        const int ch  = tid + j * 256;
        const int row = ch >> 5;        // 32 x 16B chunks per 512B source row
        const int c16 = ch & 31;
        AS1 const char* g = gbase + ((size_t)(t * 16 + row)) * (DD * 2) + c16 * 16;
        AS3 char*       l = lbase + row * (BSTR * 2) + c16 * 16;
#if HAVE_ASYNC
        __builtin_amdgcn_global_load_async_to_lds_b128(
            (AS1 v4i*)g, (AS3 v4i*)l, 0, 0);
#else
        *(AS3 v4i*)l = *(AS1 const v4i*)g;
#endif
    }
}

__device__ __forceinline__ void wait_stage_keep2() {
#if HAVE_ASYNC
    __builtin_amdgcn_s_wait_asynccnt(2);   // current tile's 2 chunks landed
#endif
}
__device__ __forceinline__ void wait_stage_all() {
#if HAVE_ASYNC
    __builtin_amdgcn_s_wait_asynccnt(0);
#endif
}

// ---------------------------------------------------------------------------
// Kernel A: L2-normalize prototypes, store f16 row-major (P x D) in workspace.
// ---------------------------------------------------------------------------
__global__ __launch_bounds__(256) void proto_normalize_kernel(
    const float* __restrict__ protos, _Float16* __restrict__ pnh) {
    const int wave = (blockIdx.x * blockDim.x + threadIdx.x) >> 5;
    const int lane = threadIdx.x & 31;
    if (wave >= PP) return;
    const float* row = protos + (size_t)wave * DD;
    float x[8];
    float ss = 0.0f;
#pragma unroll
    for (int j = 0; j < 8; ++j) {
        x[j] = row[lane + 32 * j];
        ss += x[j] * x[j];
    }
    ss = waveReduceSum32(ss);
    const float invn = 1.0f / fmaxf(sqrtf(ss), NORM_EPS);
    _Float16* orow = pnh + (size_t)wave * DD;
#pragma unroll
    for (int j = 0; j < 8; ++j)
        orow[lane + 32 * j] = (_Float16)(x[j] * invn);
}

__global__ void zero_accum_kernel(float* __restrict__ accum) {
    if (threadIdx.x < 2) accum[threadIdx.x] = 0.0f;
}

// ---------------------------------------------------------------------------
// Kernel B: 8 waves/block, 16 query rows per wave (128 rows per block).
// ---------------------------------------------------------------------------
__global__ __launch_bounds__(256) void pcl_main_kernel(
    const float* __restrict__ lv,       // (NQ, D) local vectors
    const _Float16* __restrict__ pnh,   // (P, D) normalized f16 prototypes
    const int* __restrict__ gt,         // (NQ) concepts_gt flattened
    float* __restrict__ accum)          // [0]=total, [1]=count
{
    // 64KB raw LDS: first used as per-wave A staging (8 x 8KB), then the
    // first ~17KB is reused as the double-buffered B tile store.
    __shared__ __align__(32) unsigned char smem_raw[65536];

    const int tid     = threadIdx.x;
    const int w       = tid >> 5;
    const int lane    = tid & 31;
    const int tile    = blockIdx.x * 8 + w;   // 2048 row tiles total
    const int rowbase = tile * 16;

    // Single AS3 base: all LDS accesses become ds_* (never flat_*).
    AS3 _Float16* lds16 = (AS3 _Float16*)(_Float16*)smem_raw;
    AS3 _Float16* astage = lds16 + w * 16 * DD;

    // AS1 base for the async loads (addrspace cast paid once).
    AS1 const char* gbase = (AS1 const char*)(const char*)pnh;
    AS3 char*       lbase = (AS3 char*)(char*)smem_raw;

    // --- stage 1: normalize this wave's 16 query rows into LDS as f16 ---
    for (int r = 0; r < 16; ++r) {
        const float* row = lv + (size_t)(rowbase + r) * DD;
        float x[8];
        float ss = 0.0f;
#pragma unroll
        for (int j = 0; j < 8; ++j) {
            x[j] = row[lane + 32 * j];
            ss += x[j] * x[j];
        }
        ss = waveReduceSum32(ss);
        const float invn = 1.0f / fmaxf(sqrtf(ss), NORM_EPS);
#pragma unroll
        for (int j = 0; j < 8; ++j)
            astage[r * DD + lane + 32 * j] = (_Float16)(x[j] * invn);
    }

    // --- stage 2: build A fragments (16x32 f16, contiguous k = 32*c + 16*hh + s; B uses same order) ---
    const int r  = lane & 15;
    const int hh = lane >> 4;
    v16h afrag[8];
#pragma unroll
    for (int c = 0; c < 8; ++c) {
        AS3 const _Float16* base = astage + r * DD + 32 * c + 16 * hh;
        v8h lo = *(AS3 const v8h*)(base);
        v8h hi = *(AS3 const v8h*)(base + 8);
        afrag[c] = __builtin_shufflevector(lo, hi, 0, 1, 2, 3, 4, 5, 6, 7,
                                           8, 9, 10, 11, 12, 13, 14, 15);
    }
    __syncthreads();   // astage dead; B buffers may now overwrite it

    // positive-column window start (4*k) for each of this lane's 8 C-rows
    int lo4[8];
#pragma unroll
    for (int rr = 0; rr < 8; ++rr) {
        const int q = rowbase + rr + 8 * hh;   // C layout: m = rr + 8*hh
        lo4[rr] = (q & (KC - 1)) * MM;
    }

    float s[8], pos[8];
#pragma unroll
    for (int rr = 0; rr < 8; ++rr) { s[rr] = 0.0f; pos[rr] = 0.0f; }

    const int n = lane & 15;   // C/B column held by this lane

    // --- stage 3: sweep 64 prototype tiles, double-buffered via async LDS ---
    stage_tile(gbase, lbase, 0, tid);

    for (int t = 0; t < NT; ++t) {
        const int cur = t & 1;
        if (t + 1 < NT) {
            stage_tile(gbase, lbase + (cur ^ 1) * (BUFH * 2), t + 1, tid);
            wait_stage_keep2();     // tile t's chunks complete (in-order)
        } else {
            wait_stage_all();
        }
        __syncthreads();            // all waves' staged chunks visible

        // B fragments straight from LDS (ds_load_b128 x2 per chunk).
        AS3 const _Float16* bcol = lds16 + cur * BUFH + n * BSTR + 16 * hh;
        v8f acc = {0.0f, 0.0f, 0.0f, 0.0f, 0.0f, 0.0f, 0.0f, 0.0f};
#pragma unroll
        for (int c = 0; c < 8; ++c) {
            v16h b = *(AS3 const v16h*)(bcol + 32 * c);
            acc = __builtin_amdgcn_wmma_f32_16x16x32_f16(
                false, afrag[c], false, b, (short)0, acc, false, false);
        }

        const int p = t * 16 + n;   // global prototype column index
#pragma unroll
        for (int rr = 0; rr < 8; ++rr) {
            const float v = acc[rr] * TEMP_INV;         // sims/T in [-10,10]
            s[rr] += __expf(v);                         // no shift needed
            const bool ispos = (unsigned)(p - lo4[rr]) < (unsigned)MM;
            pos[rr] += ispos ? v : 0.0f;
        }
        __syncthreads();            // done reading this buffer before t+2 overwrite
    }

    // --- stage 4: fold the 16 columns of each row (xor within half) ---
#pragma unroll
    for (int rr = 0; rr < 8; ++rr) {
#pragma unroll
        for (int off = 8; off > 0; off >>= 1) {
            s[rr]   += __shfl_xor(s[rr],   off, 32);
            pos[rr] += __shfl_xor(pos[rr], off, 32);
        }
    }

    // --- stage 5: masked accumulate (lanes n==0 hold complete rows) ---
    if (n == 0) {
        float t0 = 0.0f, c0 = 0.0f;
#pragma unroll
        for (int rr = 0; rr < 8; ++rr) {
            const int q = rowbase + rr + 8 * hh;
            if (gt[q] == 1) {
                t0 += (float)MM * __logf(s[rr]) - pos[rr];
                c0 += (float)MM;
            }
        }
        atomicAdd(&accum[0], t0);
        atomicAdd(&accum[1], c0);
    }
}

__global__ void finalize_kernel(const float* __restrict__ accum,
                                float* __restrict__ out) {
    const float total = accum[0];
    const float cnt   = accum[1];
    out[0] = (cnt > 0.0f) ? (total / fmaxf(cnt, 1.0f)) : 0.0f;
}

// ---------------------------------------------------------------------------
extern "C" void kernel_launch(void* const* d_in, const int* in_sizes, int n_in,
                              void* d_out, int out_size, void* d_ws, size_t ws_size,
                              hipStream_t stream) {
    const float* lv     = (const float*)d_in[0];  // (B,K,D) f32
    const float* protos = (const float*)d_in[1];  // (K*M,D,1,1) f32
    const int*   gt     = (const int*)d_in[2];    // (B,K) int32
    // d_in[3]: num_prototypes_per_concept (==4, baked in as MM)

    float*     out   = (float*)d_out;
    float*     accum = (float*)d_ws;                   // 2 floats
    _Float16*  pnh   = (_Float16*)((char*)d_ws + 256); // 512 KB

    zero_accum_kernel<<<1, 32, 0, stream>>>(accum);
    proto_normalize_kernel<<<PP / 8, 256, 0, stream>>>(protos, pnh);
    pcl_main_kernel<<<(NQ / 16) / 8, 256, 0, stream>>>(lv, pnh, gt, accum);
    finalize_kernel<<<1, 1, 0, stream>>>(accum, out);
}
